// MsaHmmLayer_9036611191456
// MI455X (gfx1250) — 8-bit matrix operand order repaired and checked by reading, compiled, not run on hardware
//
#include <hip/hip_runtime.h>

typedef __attribute__((ext_vector_type(16))) int   v16i;
typedef __attribute__((ext_vector_type(8)))  float v8f;

#define Mn 2
#define Bn 32
#define Ln 512
#define Qn 512
#define LQ (Ln*Qn)
#define AS 528
#define LOG_S2 11.090354888959125f

union AV { int2 d[8]; v16i v; };

static __device__ __forceinline__ unsigned int to_fp8(float x) {
    _Float16 hf = (_Float16)x;
    unsigned int h = (unsigned int)__builtin_bit_cast(unsigned short, hf);
    unsigned int t = h + 0x40u;
    unsigned int b = (t >> 7) - 64u;
    return (h < 0x23C0u) ? 0u : (b & 0xFFu);
}

static __device__ __forceinline__ int pack4_fp8(float a, float b, float c, float d) {
    return (int)(to_fp8(a) | (to_fp8(b) << 8) | (to_fp8(c) << 16) | (to_fp8(d) << 24));
}

__global__ __launch_bounds__(512, 1)
void hmm_forward_fp8(const float* __restrict__ em,
                     const float* __restrict__ A,
                     const float* __restrict__ pi,
                     float* __restrict__ out)
{
    __shared__ __align__(16) unsigned char alpha_s[Bn * AS];
    __shared__ float rowsum[2][Bn];
    __shared__ float inv_s[Bn];

    const int tid     = threadIdx.x;
    const int wid     = tid >> 5;
    const int lane    = tid & 31;
    const int lane_lo = lane & 15;
    const int hi      = lane >> 4;
    const int m       = blockIdx.x;
    const int na      = wid * 32 + lane_lo;
    const int nb      = na + 16;

    const float* Am  = A  + (size_t)m * Qn * Qn;
    const float* emm = em + (size_t)m * Bn * LQ;
    const float* pim = pi + (size_t)m * Qn;

    v16i bfrag[2][4];
    #pragma unroll
    for (int ct = 0; ct < 2; ++ct) {
        const int n = na + ct * 16;
        #pragma unroll
        for (int kc = 0; kc < 4; ++kc) {
            v16i bf;
            #pragma unroll
            for (int blk = 0; blk < 2; ++blk)
                #pragma unroll
                for (int j2 = 0; j2 < 2; ++j2)
                    #pragma unroll
                    for (int jj = 0; jj < 4; ++jj) {
                        const int q = kc * 128 + blk * 64 + j2 * 32 + (jj >> 1) * 16 + hi * 8 + (jj & 1) * 4;
                        const float* p = Am + (size_t)q * Qn + n;
                        bf[blk * 8 + j2 * 4 + jj] =
                            pack4_fp8(p[0] * 256.f, p[Qn] * 256.f,
                                      p[2 * (size_t)Qn] * 256.f, p[3 * (size_t)Qn] * 256.f);
                    }
            bfrag[ct][kc] = bf;
        }
    }

    if (tid < 64) ((float*)rowsum)[tid] = 0.f;
    __syncthreads();

    for (int idx = tid; idx < Bn * Qn; idx += 512) {
        int b = idx >> 9, q = idx & 511;
        float v = pim[q] * emm[(size_t)b * LQ + q];
        atomicAdd(&rowsum[0][b], v);
    }
    __syncthreads();
    float ll = 0.f;
    if (wid == 0) {
        float s = rowsum[0][lane];
        ll = logf(s);
        inv_s[lane] = 256.f / s;
    }
    __syncthreads();
    for (int idx = tid; idx < Bn * Qn; idx += 512) {
        int b = idx >> 9, q = idx & 511;
        float v = pim[q] * emm[(size_t)b * LQ + q];
        alpha_s[b * AS + q] = (unsigned char)to_fp8(v * inv_s[b]);
    }
    __syncthreads();

    const int abase0 = lane_lo * AS + hi * 8;
    const int abase1 = (16 + lane_lo) * AS + hi * 8;

    for (int t = 1; t < Ln; ++t) {
        const int par = t & 1;

        if (t + 1 < Ln)
            __builtin_prefetch(emm + (size_t)(tid >> 4) * LQ +
                               (size_t)(t + 1) * Qn + (size_t)(tid & 15) * 32, 0, 1);

        v8f c00 = {}, c01 = {}, c10 = {}, c11 = {};
        #pragma unroll
        for (int kc = 0; kc < 4; ++kc) {
            AV a0, a1;
            const unsigned char* p0 = alpha_s + abase0 + kc * 128;
            const unsigned char* p1 = alpha_s + abase1 + kc * 128;
            #pragma unroll
            for (int g = 0; g < 8; ++g) {
                a0.d[g] = *(const int2*)(p0 + g * 16);
                a1.d[g] = *(const int2*)(p1 + g * 16);
            }
            c00 = __builtin_amdgcn_wmma_f32_16x16x128_fp8_fp8(a0.v, bfrag[0][kc], (short)0, c00, false, false);
            c01 = __builtin_amdgcn_wmma_f32_16x16x128_fp8_fp8(a0.v, bfrag[1][kc], (short)0, c01, false, false);
            c10 = __builtin_amdgcn_wmma_f32_16x16x128_fp8_fp8(a1.v, bfrag[0][kc], (short)0, c10, false, false);
            c11 = __builtin_amdgcn_wmma_f32_16x16x128_fp8_fp8(a1.v, bfrag[1][kc], (short)0, c11, false, false);
        }

        const float* ept = emm + (size_t)t * Qn;
        #pragma unroll
        for (int i = 0; i < 8; ++i) {
            const int rA = i + 8 * hi, rB = 16 + i + 8 * hi;
            c00[i] *= ept[(size_t)rA * LQ + na];
            c01[i] *= ept[(size_t)rA * LQ + nb];
            c10[i] *= ept[(size_t)rB * LQ + na];
            c11[i] *= ept[(size_t)rB * LQ + nb];
        }

        #pragma unroll
        for (int i = 0; i < 8; ++i) {
            float s0 = c00[i] + c01[i];
            float s1 = c10[i] + c11[i];
            s0 += __shfl_xor(s0, 1); s0 += __shfl_xor(s0, 2);
            s0 += __shfl_xor(s0, 4); s0 += __shfl_xor(s0, 8);
            s1 += __shfl_xor(s1, 1); s1 += __shfl_xor(s1, 2);
            s1 += __shfl_xor(s1, 4); s1 += __shfl_xor(s1, 8);
            if (lane == 0)  { atomicAdd(&rowsum[par][i],      s0);
                              atomicAdd(&rowsum[par][16 + i], s1); }
            if (lane == 16) { atomicAdd(&rowsum[par][8 + i],  s0);
                              atomicAdd(&rowsum[par][24 + i], s1); }
        }
        __syncthreads();

        if (wid == 0) {
            float s = rowsum[par][lane];
            ll += logf(s);
            inv_s[lane] = 256.f / s;
            rowsum[1 - par][lane] = 0.f;
        }
        __syncthreads();

        #pragma unroll
        for (int i = 0; i < 8; ++i) {
            const int rA = i + 8 * hi, rB = 16 + i + 8 * hi;
            const float fA = inv_s[rA], fB = inv_s[rB];
            alpha_s[rA * AS + na] = (unsigned char)to_fp8(c00[i] * fA);
            alpha_s[rA * AS + nb] = (unsigned char)to_fp8(c01[i] * fA);
            alpha_s[rB * AS + na] = (unsigned char)to_fp8(c10[i] * fB);
            alpha_s[rB * AS + nb] = (unsigned char)to_fp8(c11[i] * fB);
        }
        __syncthreads();
    }

    if (wid == 0) out[m * Bn + lane] = ll - 511.f * LOG_S2;
}

extern "C" void kernel_launch(void* const* d_in, const int* in_sizes, int n_in,
                              void* d_out, int out_size, void* d_ws, size_t ws_size,
                              hipStream_t stream) {
    const float* em = (const float*)d_in[0];
    const float* A  = (const float*)d_in[1];
    const float* pi = (const float*)d_in[2];
    hmm_forward_fp8<<<dim3(Mn), dim3(512), 0, stream>>>(em, A, pi, (float*)d_out);
}
